// SVMLayer_24678882082903
// MI455X (gfx1250) — matrix operand order repaired and checked by reading, not compiled, not run on hardware
//
#include <hip/hip_runtime.h>
#include <hip/hip_bf16.h>

// ---------------------------------------------------------------------------
// RBF-SVM scoring layer, fused, for gfx1250 (MI455X).
//   K[i,j] = exp(-g * max(xn[i] + zn[j] - 2*X.Z^T, 0));  out[j] = b + sum_i w[i]*K[i,j]
// fp32 GEMM emulated as split-bf16 (hi + residual lo, 3 WMMA terms) on
// v_wmma_f32_16x16x32_bf16. X/Z are pre-split to bf16 planes ONCE so the hot
// loop is pure loads + WMMA (no conversion VALU). X planes are stored with a
// k-permutation (swap bits 3<->4 within each 32-k block) so each lane's A
// fragment is one contiguous 32B read. 2x2 tile register blocking.
// ---------------------------------------------------------------------------

#define FEAT_D   512
#define NCHUNKS  16
#define GAMMA    (1.0f / 512.0f)
#define NEG_G_LOG2E (-1.4426950408889634f * GAMMA)

typedef __attribute__((ext_vector_type(16))) __bf16 v16bf;
typedef __attribute__((ext_vector_type(8)))  __bf16 v8bf;
typedef __attribute__((ext_vector_type(8)))  float  v8f;

#define WMMA_BF16(A, B, C) \
  __builtin_amdgcn_wmma_f32_16x16x32_bf16(false, (A), false, (B), (short)0, (C), false, false)

// ---------------------------------------------------------------------------
// Kernel 1: row norms of X and Z, and w = a*Y. One wave (32 lanes) per row.
// ---------------------------------------------------------------------------
__global__ __launch_bounds__(128) void svm_prep(
    const float* __restrict__ Z, const float* __restrict__ X,
    const float* __restrict__ Y, const float* __restrict__ a,
    float* __restrict__ xn, float* __restrict__ zn, float* __restrict__ w,
    int N, int M) {
  const int lane = threadIdx.x & 31;
  const int wave = threadIdx.x >> 5;
  const int row  = blockIdx.x * 4 + wave;
  if (row >= N + M) return;

  const float* src = (row < N) ? (X + (size_t)row * FEAT_D)
                               : (Z + (size_t)(row - N) * FEAT_D);
  float s = 0.0f;
  for (int k = lane * 4; k < FEAT_D; k += 32 * 4) {
    float4 v = *(const float4*)(src + k);
    s = fmaf(v.x, v.x, s); s = fmaf(v.y, v.y, s);
    s = fmaf(v.z, v.z, s); s = fmaf(v.w, v.w, s);
  }
#pragma unroll
  for (int off = 16; off > 0; off >>= 1) s += __shfl_xor(s, off, 32);
  if (lane == 0) {
    if (row < N) { xn[row] = s; w[row] = a[row] * Y[row]; }
    else         { zn[row - N] = s; }
  }
}

// ---------------------------------------------------------------------------
// Kernel 2: split fp32 -> bf16 hi + residual lo planes (done once).
// permuteA=1 stores with k bits 3<->4 swapped per 32-block so a lane's A
// fragment {k+half*8+0..7, k+16+half*8+0..7} becomes contiguous 32B.
// Each thread converts 8 consecutive source elements (one aligned group).
// ---------------------------------------------------------------------------
__global__ __launch_bounds__(256) void svm_split(
    const float* __restrict__ src, __bf16* __restrict__ hi,
    __bf16* __restrict__ lo, int rows, int permuteA) {
  size_t idx = ((size_t)blockIdx.x * blockDim.x + threadIdx.x) * 8;
  if (idx >= (size_t)rows * FEAT_D) return;

  size_t dst = idx;
  if (permuteA) {
    size_t row = idx >> 9;                 // /FEAT_D
    int kk = (int)(idx & 511);
    int g  = (kk >> 3) & 3;                // 8-element group within 32-block
    int gp = ((g & 1) << 1) | (g >> 1);    // swap bit3<->bit4 of k
    dst = (row << 9) + (size_t)((kk & ~31) | (gp << 3));
  }

  float4 v0 = *(const float4*)(src + idx);
  float4 v1 = *(const float4*)(src + idx + 4);
  float f[8] = {v0.x, v0.y, v0.z, v0.w, v1.x, v1.y, v1.z, v1.w};
  v8bf h, l;
#pragma unroll
  for (int e = 0; e < 8; ++e) {
    __bf16 hb = (__bf16)f[e];
    h[e] = hb;
    l[e] = (__bf16)(f[e] - (float)hb);
  }
  *(v8bf*)(hi + dst) = h;
  *(v8bf*)(lo + dst) = l;
}

// ---------------------------------------------------------------------------
// Weighted-exp reduction of one fp32 C/D fragment into a column partial.
// ---------------------------------------------------------------------------
__device__ __forceinline__ float tile_reduce(const v8f& acc,
                                             const float* __restrict__ xnr,
                                             const float* __restrict__ wr,
                                             float znj, float s) {
#pragma unroll
  for (int r = 0; r < 8; ++r) {
    float sq = fmaxf(xnr[r] + znj - 2.0f * acc[r], 0.0f);
    float kv = exp2f(sq * NEG_G_LOG2E);        // native v_exp_f32
    s = fmaf(wr[r], kv, s);
  }
  return s;
}

// ---------------------------------------------------------------------------
// Kernel 3: fused WMMA GEMM + exp + weighted column reduction.
// One wave per (32-column j-block, N-chunk); 2 i-tiles x 2 j-tiles register
// blocking: 8 x 32B fragment loads feed 12 WMMAs per k-step.
// C/D layout: vgpr r, lane L -> row r + 8*(L/16), col L%16.
// ---------------------------------------------------------------------------
__global__ __launch_bounds__(128) void svm_main(
    const __bf16* __restrict__ Zhi, const __bf16* __restrict__ Zlo,
    const __bf16* __restrict__ Xhi, const __bf16* __restrict__ Xlo,
    const float* __restrict__ xn, const float* __restrict__ zn,
    const float* __restrict__ w, float* __restrict__ partial,
    int N, int M) {
  const int lane  = threadIdx.x & 31;
  const int wave  = threadIdx.x >> 5;
  const int task  = blockIdx.x * 4 + wave;
  const int jBlocks = M >> 5;
  if (task >= jBlocks * NCHUNKS) return;

  const int jb    = task % jBlocks;
  const int chunk = task / jBlocks;
  const int rowsPerChunk = N / NCHUNKS;    // 512 for N=8192
  const int i0base = chunk * rowsPerChunk;

  const int col  = lane & 15;
  const int half = lane >> 4;
  const int j0   = jb * 32 + col;
  const int j1   = j0 + 16;

  const __bf16* Zh0 = Zhi + (size_t)j0 * FEAT_D;
  const __bf16* Zl0 = Zlo + (size_t)j0 * FEAT_D;
  const __bf16* Zh1 = Zhi + (size_t)j1 * FEAT_D;
  const __bf16* Zl1 = Zlo + (size_t)j1 * FEAT_D;
  const float zn0 = zn[j0], zn1 = zn[j1];

  float sum0 = 0.0f, sum1 = 0.0f;

  for (int it = 0; it < rowsPerChunk; it += 32) {
    const int i0 = i0base + it;
    const __bf16* Xh0 = Xhi + (size_t)(i0 + col) * FEAT_D;
    const __bf16* Xl0 = Xlo + (size_t)(i0 + col) * FEAT_D;
    const __bf16* Xh1 = Xhi + (size_t)(i0 + 16 + col) * FEAT_D;
    const __bf16* Xl1 = Xlo + (size_t)(i0 + 16 + col) * FEAT_D;

    if (it + 32 < rowsPerChunk) {          // global_prefetch_b8 hints
      __builtin_prefetch(Xh0 + 32 * FEAT_D, 0, 1);
      __builtin_prefetch(Xh1 + 32 * FEAT_D, 0, 1);
    }

    v8f acc00 = {}, acc01 = {}, acc10 = {}, acc11 = {};
#pragma unroll 2
    for (int k = 0; k < FEAT_D; k += 32) {
      const int off = k + half * 16;       // 32B-aligned fragment offset
      v16bf ah0 = *(const v16bf*)(Xh0 + off);
      v16bf al0 = *(const v16bf*)(Xl0 + off);
      v16bf ah1 = *(const v16bf*)(Xh1 + off);
      v16bf al1 = *(const v16bf*)(Xl1 + off);
      v16bf bh0 = *(const v16bf*)(Zh0 + off);
      v16bf bl0 = *(const v16bf*)(Zl0 + off);
      v16bf bh1 = *(const v16bf*)(Zh1 + off);
      v16bf bl1 = *(const v16bf*)(Zl1 + off);

      // Interleave independent accumulators to hide WMMA D->C latency.
      acc00 = WMMA_BF16(ah0, bh0, acc00);
      acc01 = WMMA_BF16(ah0, bh1, acc01);
      acc10 = WMMA_BF16(ah1, bh0, acc10);
      acc11 = WMMA_BF16(ah1, bh1, acc11);
      acc00 = WMMA_BF16(ah0, bl0, acc00);
      acc01 = WMMA_BF16(ah0, bl1, acc01);
      acc10 = WMMA_BF16(ah1, bl0, acc10);
      acc11 = WMMA_BF16(ah1, bl1, acc11);
      acc00 = WMMA_BF16(al0, bh0, acc00);
      acc01 = WMMA_BF16(al0, bh1, acc01);
      acc10 = WMMA_BF16(al1, bh0, acc10);
      acc11 = WMMA_BF16(al1, bh1, acc11);
    }

    // Epilogue: rows i0 + half*8 + r (tile 0) and +16 (tile 1).
    float xnr0[8], wr0[8], xnr1[8], wr1[8];
    const float* xp0 = xn + i0 + half * 8;
    const float* wp0 = w  + i0 + half * 8;
    ((float4*)xnr0)[0] = ((const float4*)xp0)[0];
    ((float4*)xnr0)[1] = ((const float4*)xp0)[1];
    ((float4*)wr0)[0]  = ((const float4*)wp0)[0];
    ((float4*)wr0)[1]  = ((const float4*)wp0)[1];
    ((float4*)xnr1)[0] = ((const float4*)(xp0 + 16))[0];
    ((float4*)xnr1)[1] = ((const float4*)(xp0 + 16))[1];
    ((float4*)wr1)[0]  = ((const float4*)(wp0 + 16))[0];
    ((float4*)wr1)[1]  = ((const float4*)(wp0 + 16))[1];

    sum0 = tile_reduce(acc00, xnr0, wr0, zn0, sum0);
    sum1 = tile_reduce(acc01, xnr0, wr0, zn1, sum1);
    sum0 = tile_reduce(acc10, xnr1, wr1, zn0, sum0);
    sum1 = tile_reduce(acc11, xnr1, wr1, zn1, sum1);
  }

  // Lane L and L+16 hold row-halves of the same column -> combine (wave32).
  sum0 += __shfl_xor(sum0, 16, 32);
  sum1 += __shfl_xor(sum1, 16, 32);
  if (half == 0) {
    partial[(size_t)chunk * M + j0] = sum0;
    partial[(size_t)chunk * M + j1] = sum1;
  }
}

// ---------------------------------------------------------------------------
// Kernel 4: out[j] = b + sum_chunks partial[c][j]
// ---------------------------------------------------------------------------
__global__ void svm_finalize(const float* __restrict__ partial,
                             const float* __restrict__ b,
                             float* __restrict__ out, int M) {
  int jidx = blockIdx.x * blockDim.x + threadIdx.x;
  if (jidx >= M) return;
  float s = b[0];
#pragma unroll
  for (int c = 0; c < NCHUNKS; ++c) s += partial[(size_t)c * M + jidx];
  out[jidx] = s;
}

extern "C" void kernel_launch(void* const* d_in, const int* in_sizes, int n_in,
                              void* d_out, int out_size, void* d_ws, size_t ws_size,
                              hipStream_t stream) {
  const float* Z = (const float*)d_in[0];   // inputs [M, d]
  const float* X = (const float*)d_in[1];   // support vectors [N, d]
  const float* Y = (const float*)d_in[2];   // labels [N, 1]
  const float* a = (const float*)d_in[3];   // alphas [N, 1]
  const float* b = (const float*)d_in[4];   // bias [1, 1]
  const int M = in_sizes[0] / FEAT_D;       // 4096
  const int N = in_sizes[1] / FEAT_D;       // 8192

  // Workspace layout (fp32 region is 1KB-multiple so bf16 planes stay aligned)
  float* wsf     = (float*)d_ws;
  float* xn      = wsf;                     // [N]
  float* zn      = xn + N;                  // [M]
  float* w       = zn + M;                  // [N]
  float* partial = w + N;                   // [NCHUNKS, M]
  __bf16* Xhi = (__bf16*)(partial + (size_t)NCHUNKS * M);
  __bf16* Xlo = Xhi + (size_t)N * FEAT_D;
  __bf16* Zhi = Xlo + (size_t)N * FEAT_D;
  __bf16* Zlo = Zhi + (size_t)M * FEAT_D;

  const int rows = N + M;
  svm_prep<<<(rows + 3) / 4, 128, 0, stream>>>(Z, X, Y, a, xn, zn, w, N, M);

  const int tX = (N * FEAT_D) / 8, tZ = (M * FEAT_D) / 8;
  svm_split<<<(tX + 255) / 256, 256, 0, stream>>>(X, Xhi, Xlo, N, /*permuteA=*/1);
  svm_split<<<(tZ + 255) / 256, 256, 0, stream>>>(Z, Zhi, Zlo, M, /*permuteA=*/1);

  const int tasks = (M >> 5) * NCHUNKS;     // one wave per (j-block, N-chunk)
  svm_main<<<(tasks + 3) / 4, 128, 0, stream>>>(Zhi, Zlo, Xhi, Xlo, xn, zn, w,
                                                partial, N, M);

  svm_finalize<<<(M + 255) / 256, 256, 0, stream>>>(partial, b, (float*)d_out, M);
}
